// ComplexSSMLayer_6657199309143
// MI455X (gfx1250) — matrix operand order repaired and checked by reading, not compiled, not run on hardware
//
#include <hip/hip_runtime.h>
#include <hip/hip_bf16.h>

// ---------------------------------------------------------------------------
// Complex SSM layer for gfx1250 (MI455X, wave32, WMMA).
// All GEMMs use v_wmma_f32_16x16x32_f16 (f16 in, f32 accumulate).
// Global->LDS staging uses GLOBAL_LOAD_ASYNC_TO_LDS_B128 when available.
// ---------------------------------------------------------------------------

#define B_  4
#define L_  4096
#define DIM_ 2048
#define S_  128

typedef __attribute__((ext_vector_type(16))) _Float16 v16h;
typedef __attribute__((ext_vector_type(8)))  float    v8f;

#define WMMA_F16(a, b, c) \
  __builtin_amdgcn_wmma_f32_16x16x32_f16(false, (a), false, (b), (short)0, (c), false, false)

#if __has_builtin(__builtin_amdgcn_global_load_async_to_lds_b128)
#define HAVE_ASYNC_LDS 1
// The builtin takes (v4i addrspace(1)*, v4i addrspace(3)*, imm offset, imm cpol)
typedef int v4i_ __attribute__((vector_size(16)));
typedef __attribute__((address_space(1))) v4i_ gv4i_t;
typedef __attribute__((address_space(3))) v4i_ lv4i_t;
#else
#define HAVE_ASYNC_LDS 0
#endif

__device__ __forceinline__ void wait_async_lds() {
#if __has_builtin(__builtin_amdgcn_s_wait_asynccnt)
  __builtin_amdgcn_s_wait_asynccnt(0);
#else
  asm volatile("s_wait_asynccnt 0x0" ::: "memory");
#endif
}

__device__ __forceinline__ float fsqrt_fast(float x) {
#if __has_builtin(__builtin_amdgcn_sqrtf)
  return __builtin_amdgcn_sqrtf(x);   // raw v_sqrt_f32, no precise fixup
#else
  return sqrtf(x);
#endif
}

__device__ __forceinline__ float frcp_fast(float x) {
#if __has_builtin(__builtin_amdgcn_rcpf)
  return __builtin_amdgcn_rcpf(x);    // raw v_rcp_f32
#else
  return 1.0f / x;
#endif
}

// Fragment-order packing: flat idx -> (K, n) of a (K x N) f16 weight matrix.
// Layout: [K/32 ktiles][N/16 ntiles][32 lanes][16 halves]; each lane's
// B-fragment is one contiguous 32-byte v16h.
// B-matrix 32x16 layout: half h = lane>>4, elem e -> K = e + 8*((e>>3)+h)
// (e<8: K=8h+e; e>=8: K=16+8h+(e-8)); column n = lane & 15.
__device__ __forceinline__ void frag_pos(int idx, int NT, int& K, int& n) {
  int e    = idx & 15;
  int lane = (idx >> 4) & 31;
  int nt   = (idx >> 9) % NT;
  int kt   = (idx >> 9) / NT;
  K = kt * 32 + e + 8 * ((e >> 3) + (lane >> 4));
  n = nt * 16 + (lane & 15);
}

// ---------------------------------------------------------------------------
// Kernel 1: pack all weights into fragment-order f16 buffers.
//   Wdt : (4096 x 128)  = dt_w^T                 (x_flat -> dt logits)
//   Wb  : (4096 x 256)  = [[Br, Bi], [-Bi, Br]]  (x_flat -> Bx_r | Bx_i)
//   Wpg : (2048 x 128)  = pg_w^T                 (|x|    -> gate logits)
//   Wcr : (256  x 2048) = [[Cr], [-Ci]]          (h_flat -> y_r)
//   Wci : (256  x 2048) = [[Ci], [ Cr]]          (h_flat -> y_i)
// ---------------------------------------------------------------------------
#define CNT_WDT (4096 * 128)
#define CNT_WB  (4096 * 256)
#define CNT_WPG (2048 * 128)
#define CNT_WC  (256 * 2048)
#define CNT_ALL (CNT_WDT + CNT_WB + CNT_WPG + 2 * CNT_WC)

__global__ __launch_bounds__(256) void pack_kernel(
    const float* __restrict__ dt_w, const float* __restrict__ Br,
    const float* __restrict__ Bi,   const float* __restrict__ Cr,
    const float* __restrict__ Ci,   const float* __restrict__ pg_w,
    _Float16* __restrict__ Wdt, _Float16* __restrict__ Wb,
    _Float16* __restrict__ Wpg, _Float16* __restrict__ Wcr,
    _Float16* __restrict__ Wci) {
  int tid = blockIdx.x * blockDim.x + threadIdx.x;
  if (tid >= CNT_ALL) return;
  int K, n;
  if (tid < CNT_WDT) {
    int idx = tid;
    frag_pos(idx, 8, K, n);                       // K in [0,4096), n in [0,128)
    Wdt[idx] = (_Float16)dt_w[n * 4096 + K];      // dt_w is (S, 2*DIM)
  } else if (tid < CNT_WDT + CNT_WB) {
    int idx = tid - CNT_WDT;
    frag_pos(idx, 16, K, n);                      // K in [0,4096), n in [0,256)
    float v;
    if (K < DIM_) {                               // xr rows: [Br | Bi]
      v = (n < S_) ? Br[K * S_ + n] : Bi[K * S_ + (n - S_)];
    } else {                                      // xi rows: [-Bi | Br]
      int d = K - DIM_;
      v = (n < S_) ? -Bi[d * S_ + n] : Br[d * S_ + (n - S_)];
    }
    Wb[idx] = (_Float16)v;
  } else if (tid < CNT_WDT + CNT_WB + CNT_WPG) {
    int idx = tid - (CNT_WDT + CNT_WB);
    frag_pos(idx, 8, K, n);                       // K in [0,2048), n in [0,128)
    Wpg[idx] = (_Float16)pg_w[n * DIM_ + K];      // pg_w is (S, DIM)
  } else if (tid < CNT_WDT + CNT_WB + CNT_WPG + CNT_WC) {
    int idx = tid - (CNT_WDT + CNT_WB + CNT_WPG);
    frag_pos(idx, 128, K, n);                     // K in [0,256), n in [0,2048)
    float v = (K < S_) ? Cr[K * DIM_ + n] : -Ci[(K - S_) * DIM_ + n];
    Wcr[idx] = (_Float16)v;
  } else {
    int idx = tid - (CNT_WDT + CNT_WB + CNT_WPG + CNT_WC);
    frag_pos(idx, 128, K, n);
    float v = (K < S_) ? Ci[K * DIM_ + n] : Cr[(K - S_) * DIM_ + n];
    Wci[idx] = (_Float16)v;
  }
}

// ---------------------------------------------------------------------------
// Kernel 2: fused projections. One 16-row tile per workgroup, 8 waves; wave w
// owns state columns [w*16, w*16+16). Streams K (dim) in 64-wide stages
// (2 k-tiles) through LDS via async global->LDS DMA; 14 wmmas per wave per
// stage. Epilogue writes interleaved {Ar,Ai,Bxr,Bxi} float4 per (b,l,s).
// ---------------------------------------------------------------------------
__global__ __launch_bounds__(256) void proj_kernel(
    const float* __restrict__ x, const float* __restrict__ dt_b,
    const float* __restrict__ dt_bias, const float* __restrict__ lAr,
    const float* __restrict__ lAi, const float* __restrict__ pg_b,
    const _Float16* __restrict__ Wdt, const _Float16* __restrict__ Wb,
    const _Float16* __restrict__ Wpg, float4* __restrict__ ABx) {
  __shared__ __align__(16) float xs[16][128];  // 16 rows x 64 k x {re,im}

  const int tile = blockIdx.x;
  const int row0 = tile * 16;
  const int b    = row0 / L_;
  const int l0   = row0 % L_;
  const int t    = threadIdx.x;
  const int w    = t >> 5;
  const int lane = t & 31;
  const int half = lane >> 4;
  const int mcol = lane & 15;  // A-frag row / C-frag column

  v8f acc_dt = {}, acc_pg = {}, acc_br = {}, acc_bi = {};
  const v16h* WdtF = (const v16h*)Wdt;  // [kt][8 nt][32 lanes]
  const v16h* WbF  = (const v16h*)Wb;   // [kt][16 nt][32 lanes]
  const v16h* WpgF = (const v16h*)Wpg;  // [kt][8 nt][32 lanes]

  for (int k0 = 0; k0 < DIM_; k0 += 64) {
    __syncthreads();  // previous stage fully consumed
    // Stage 16 rows x 128 f32 (= 512 float4): 2 x b128 per thread.
#pragma unroll
    for (int j = 0; j < 2; ++j) {
      int fi4 = t + j * 256;
      int r = fi4 >> 5;       // 32 float4 per row
      int p = fi4 & 31;
      const float* gsrc =
          x + ((size_t)(b * L_ + l0 + r) * DIM_ + k0) * 2 + p * 4;
      float* ldst = &xs[r][p * 4];
#if HAVE_ASYNC_LDS
      __builtin_amdgcn_global_load_async_to_lds_b128(
          (gv4i_t*)gsrc, (lv4i_t*)ldst, 0, 0);
#else
      *(float4*)ldst = *(const float4*)gsrc;
#endif
    }
#if HAVE_ASYNC_LDS
    wait_async_lds();
#endif
    __syncthreads();

#pragma unroll
    for (int su = 0; su < 2; ++su) {
      // A-fragments (16x32 f16): elem e -> K = e + 8*((e>>3)+half)
      v16h fr, fim, fc;
#pragma unroll
      for (int e = 0; e < 16; ++e) {
        int kk = su * 32 + e + 8 * ((e >> 3) + half);
        float xr = xs[mcol][kk * 2 + 0];
        float xi = xs[mcol][kk * 2 + 1];
        fr[e]  = (_Float16)xr;
        fim[e] = (_Float16)xi;
        fc[e]  = (_Float16)fsqrt_fast(xr * xr + xi * xi);
      }
      const int ktr = (k0 >> 5) + su;           // k-tile in the xr block
      const int kti = ((DIM_ + k0) >> 5) + su;  // k-tile in the xi block

      v16h wdt_r = WdtF[(ktr * 8 + w) * 32 + lane];
      v16h wdt_i = WdtF[(kti * 8 + w) * 32 + lane];
      v16h wb_rr = WbF[(ktr * 16 + w) * 32 + lane];      //  Br  -> Bx_r
      v16h wb_ri = WbF[(ktr * 16 + 8 + w) * 32 + lane];  //  Bi  -> Bx_i
      v16h wb_ir = WbF[(kti * 16 + w) * 32 + lane];      // -Bi  -> Bx_r
      v16h wb_ii = WbF[(kti * 16 + 8 + w) * 32 + lane];  //  Br  -> Bx_i
      v16h wpg   = WpgF[(ktr * 8 + w) * 32 + lane];

      acc_dt = WMMA_F16(fr,  wdt_r, acc_dt);
      acc_dt = WMMA_F16(fim, wdt_i, acc_dt);
      acc_br = WMMA_F16(fr,  wb_rr, acc_br);
      acc_br = WMMA_F16(fim, wb_ir, acc_br);
      acc_bi = WMMA_F16(fr,  wb_ri, acc_bi);
      acc_bi = WMMA_F16(fim, wb_ii, acc_bi);
      acc_pg = WMMA_F16(fc,  wpg,   acc_pg);
    }
  }

  // Epilogue: softplus / multi-timescale decay / gated state protection.
  const int s = w * 16 + mcol;
  const float dtb = dt_b[s] + dt_bias[s];
  const float ear = __expf(lAr[s]);
  const float cv = __cosf(lAi[s]), sv = __sinf(lAi[s]);
  const float pgb = pg_b[s];
#pragma unroll
  for (int v = 0; v < 8; ++v) {
    int m = v + half * 8;
    size_t idx = (size_t)(b * L_ + l0 + m) * S_ + s;
    float tdt  = acc_dt[v] + dtb;
    float dt   = (tdt > 20.f) ? tdt : __logf(1.f + __expf(tdt));
    float amag = __expf(-dt * ear);
    float arv = amag * cv, aiv = amag * sv;
    float p = frcp_fast(1.f + __expf(-(acc_pg[v] + pgb)));
    float q = 1.f - p;
    float sc = dt * q;
    ABx[idx] = make_float4(p + q * arv, q * aiv, acc_br[v] * sc,
                           acc_bi[v] * sc);
  }
}

// ---------------------------------------------------------------------------
// Kernel 3: chunked complex linear scan h[t] = A[t]*h[t-1] + Bx[t].
// Block = 512 threads = 16 state lanes x 32 chunks of length 128.
// Pass 1: in-chunk compose; LDS exclusive scan of chunk summaries; Pass 2:
// apply, writing h (B,L,S) float2 and h_last.
// ---------------------------------------------------------------------------
__global__ __launch_bounds__(512) void scan_kernel(
    const float4* __restrict__ ABx, float2* __restrict__ h2,
    float2* __restrict__ hlast) {
  const int CL = L_ / 32;  // 32 chunks x 128
  __shared__ float sar[16 * 32], sai[16 * 32], sbr[16 * 32], sbi[16 * 32];

  const int t = threadIdx.x;
  const int sIdx = t & 15;
  const int chunk = t >> 4;
  const int b  = blockIdx.x / (S_ / 16);
  const int sg = blockIdx.x % (S_ / 16);
  const int s  = sg * 16 + sIdx;
  const size_t base = (size_t)b * L_ * S_ + s;
  const int l0 = chunk * CL;

  // Pass 1: compose chunk transform.
  float ar = 1.f, ai = 0.f, br = 0.f, bi = 0.f;
  for (int j = 0; j < CL; ++j) {
    float4 v = ABx[base + (size_t)(l0 + j) * S_];
    float nar = v.x * ar - v.y * ai, nai = v.x * ai + v.y * ar;
    float nbr = v.x * br - v.y * bi + v.z, nbi = v.x * bi + v.y * br + v.w;
    ar = nar; ai = nai; br = nbr; bi = nbi;
  }
  int o = sIdx * 32 + chunk;
  sar[o] = ar; sai[o] = ai; sbr[o] = br; sbi[o] = bi;
  __syncthreads();

  // Exclusive scan of the 32 chunk summaries per state lane.
  if (t < 16) {
    float pa_r = 1.f, pa_i = 0.f, pb_r = 0.f, pb_i = 0.f;
    for (int c = 0; c < 32; ++c) {
      int oo = t * 32 + c;
      float ca_r = sar[oo], ca_i = sai[oo], cb_r = sbr[oo], cb_i = sbi[oo];
      sar[oo] = pa_r; sai[oo] = pa_i; sbr[oo] = pb_r; sbi[oo] = pb_i;
      float na_r = ca_r * pa_r - ca_i * pa_i;
      float na_i = ca_r * pa_i + ca_i * pa_r;
      float nb_r = ca_r * pb_r - ca_i * pb_i + cb_r;
      float nb_i = ca_r * pb_i + ca_i * pb_r + cb_i;
      pa_r = na_r; pa_i = na_i; pb_r = nb_r; pb_i = nb_i;
    }
  }
  __syncthreads();

  // Pass 2: apply. Incoming state = b-part of the exclusive prefix.
  float hpr = sbr[o], hpi = sbi[o];
  for (int j = 0; j < CL; ++j) {
    int l = l0 + j;
    size_t idx = base + (size_t)l * S_;
    float4 v = ABx[idx];
    float h_r = v.x * hpr - v.y * hpi + v.z;
    float h_i = v.x * hpi + v.y * hpr + v.w;
    h2[idx] = make_float2(h_r, h_i);
    hpr = h_r; hpi = h_i;
    if (l == L_ - 1) hlast[b * S_ + s] = make_float2(hpr, hpi);
  }
}

// ---------------------------------------------------------------------------
// Kernel 4: y = h @ [[Cr,-Ci];[Ci,Cr]] + D*x (complex). One 16-row tile per
// workgroup; A-fragments (h_flat, K=256) cached in VGPRs; 16 N-tiles/wave x 8
// K-steps x {real,imag} wmmas; fused interleaved output write.
// ---------------------------------------------------------------------------
__global__ __launch_bounds__(256) void out_kernel(
    const float2* __restrict__ h2, const _Float16* __restrict__ Wcr,
    const _Float16* __restrict__ Wci, const float* __restrict__ x,
    const float* __restrict__ Dv, float* __restrict__ y) {
  __shared__ __align__(16) float hs[16][256];

  const int tile = blockIdx.x;
  const int row0 = tile * 16;
  const int b    = row0 / L_;
  const int l0   = row0 % L_;
  const int t    = threadIdx.x;
  const int w    = t >> 5;
  const int lane = t & 31;
  const int half = lane >> 4;

  // Stage h_flat tile (16 rows x [hr(128) | hi(128)]) in LDS.
#pragma unroll
  for (int i = 0; i < 8; ++i) {
    int e2 = t + i * 256;  // 2048 float2 total
    int r = e2 >> 7;
    int s = e2 & 127;
    float2 v = h2[(size_t)(b * L_ + l0 + r) * S_ + s];
    hs[r][s] = v.x;
    hs[r][128 + s] = v.y;
  }
  __syncthreads();

  // Build all 8 K-chunk A-fragments once.
  v16h af[8];
  const int arow = lane & 15;
#pragma unroll
  for (int kt = 0; kt < 8; ++kt) {
#pragma unroll
    for (int e = 0; e < 16; ++e) {
      int K = kt * 32 + e + 8 * ((e >> 3) + half);
      af[kt][e] = (_Float16)hs[arow][K];
    }
  }

  const v16h* WcrF = (const v16h*)Wcr;  // [kt][128 nt][32 lanes]
  const v16h* WciF = (const v16h*)Wci;

  for (int ntl = 0; ntl < 16; ++ntl) {
    int nt = ntl * 8 + w;
    v8f yr_acc = {}, yi_acc = {};
#pragma unroll
    for (int kt = 0; kt < 8; ++kt) {
      v16h bf_r = WcrF[(kt * 128 + nt) * 32 + lane];
      v16h bf_i = WciF[(kt * 128 + nt) * 32 + lane];
      yr_acc = WMMA_F16(af[kt], bf_r, yr_acc);
      yi_acc = WMMA_F16(af[kt], bf_i, yi_acc);
    }
    int col = nt * 16 + (lane & 15);
    float2 Dc = ((const float2*)Dv)[col];
#pragma unroll
    for (int v = 0; v < 8; ++v) {
      int m = v + half * 8;
      size_t rg = (size_t)(b * L_ + l0 + m);
      size_t eidx = rg * DIM_ + col;
      float2 xc = ((const float2*)x)[eidx];
      ((float2*)y)[eidx] =
          make_float2(yr_acc[v] + Dc.x * xc.x - Dc.y * xc.y,
                      yi_acc[v] + Dc.x * xc.y + Dc.y * xc.x);
    }
  }
}

// ---------------------------------------------------------------------------
// Host launcher.
// ---------------------------------------------------------------------------
extern "C" void kernel_launch(void* const* d_in, const int* in_sizes, int n_in,
                              void* d_out, int out_size, void* d_ws,
                              size_t ws_size, hipStream_t stream) {
  // setup_inputs order:
  const float* x        = (const float*)d_in[0];   // (B,L,DIM,2)
  const float* lAr      = (const float*)d_in[1];   // (S,)
  const float* lAi      = (const float*)d_in[2];   // (S,)
  const float* Dv       = (const float*)d_in[3];   // (DIM,2)
  const float* dt_w     = (const float*)d_in[4];   // (S, 2*DIM)
  const float* dt_b     = (const float*)d_in[5];   // (S,)
  const float* dt_bias  = (const float*)d_in[6];   // (S,)
  const float* Br       = (const float*)d_in[7];   // (DIM,S)
  const float* Bi       = (const float*)d_in[8];   // (DIM,S)
  const float* Cr       = (const float*)d_in[9];   // (S,DIM)
  const float* Ci       = (const float*)d_in[10];  // (S,DIM)
  const float* pg_w     = (const float*)d_in[11];  // (S,DIM)
  const float* pg_b     = (const float*)d_in[12];  // (S,)

  float*  y_out = (float*)d_out;                                  // (B,L,DIM,2)
  float2* hlast = (float2*)((float*)d_out + (size_t)B_ * L_ * DIM_ * 2);

  // Workspace carve-out (256-byte aligned regions).
  size_t off = 0;
  char* ws = (char*)d_ws;
  auto carve = [&](size_t bytes) -> void* {
    void* p = ws + off;
    off += (bytes + 255) & ~(size_t)255;
    return p;
  };
  _Float16* Wdt = (_Float16*)carve((size_t)CNT_WDT * 2);
  _Float16* Wb  = (_Float16*)carve((size_t)CNT_WB * 2);
  _Float16* Wpg = (_Float16*)carve((size_t)CNT_WPG * 2);
  _Float16* Wcr = (_Float16*)carve((size_t)CNT_WC * 2);
  _Float16* Wci = (_Float16*)carve((size_t)CNT_WC * 2);
  const size_t nBLS = (size_t)B_ * L_ * S_;
  float4* ABx = (float4*)carve(nBLS * sizeof(float4));
  float2* h2  = (float2*)carve(nBLS * sizeof(float2));

  // 1) Pack weights into fragment-order f16.
  pack_kernel<<<(CNT_ALL + 255) / 256, 256, 0, stream>>>(
      dt_w, Br, Bi, Cr, Ci, pg_w, Wdt, Wb, Wpg, Wcr, Wci);

  // 2) Fused projections (dt / B-proj / protect-gate) -> interleaved A,Bx.
  proj_kernel<<<(B_ * L_) / 16, 256, 0, stream>>>(
      x, dt_b, dt_bias, lAr, lAi, pg_b, Wdt, Wb, Wpg, ABx);

  // 3) Chunked complex scan over L.
  scan_kernel<<<B_ * (S_ / 16), 512, 0, stream>>>(ABx, h2, hlast);

  // 4) Output projection + D skip.
  out_kernel<<<(B_ * L_) / 16, 256, 0, stream>>>(h2, Wcr, Wci, x, Dv, y_out);
}
